// GeneralAttention_22531398434933
// MI455X (gfx1250) — matrix operand order repaired and checked by reading, not compiled, not run on hardware
//
#include <hip/hip_runtime.h>

// ---------------------------------------------------------------------------
// GeneralAttention for MI455X (gfx1250, wave32, WMMA)
//   q_proj = Q @ W^T + b ; scoresT = K @ q_proj^T ; softmax rows (mask->-inf)
//   context = weights @ V.  GEMMs in bf16 WMMA with fp32 accumulate.
//   Global->LDS staging: GLOBAL_LOAD_ASYNC_TO_LDS_B128 (ASYNCcnt),
//   3-deep tile pipeline with incremental addressing + peeled drain.
// ---------------------------------------------------------------------------

typedef __attribute__((ext_vector_type(16))) __bf16       v16bf;
typedef __attribute__((ext_vector_type(8)))  float        v8f;
typedef __attribute__((ext_vector_type(4)))  unsigned int v4u;
typedef __attribute__((ext_vector_type(2)))  unsigned int v2u;
typedef __attribute__((ext_vector_type(4)))  float        v4f;
typedef unsigned short ushort_t;

#define B_  8
#define S1_ 2048
#define S2_ 2048
#define H_  1024

__device__ __forceinline__ ushort_t f32_bf16(float f) {
  union { float f; unsigned int u; } c; c.f = f;
  unsigned int x = c.u;
  x += 0x7fffu + ((x >> 16) & 1u);           // round-to-nearest-even
  return (ushort_t)(x >> 16);
}

// ---------------------------------------------------------------------------
// fp32 -> bf16 elementwise convert (vectorized x4, grid-stride)
// ---------------------------------------------------------------------------
__global__ void cvt_f32_bf16_kernel(const float* __restrict__ src,
                                    ushort_t* __restrict__ dst, long long n4) {
  long long i      = (long long)blockIdx.x * blockDim.x + threadIdx.x;
  long long stride = (long long)gridDim.x * blockDim.x;
  for (long long j = i; j < n4; j += stride) {
    v4f v = ((const v4f*)src)[j];
    v2u o;
    o.x = (unsigned int)f32_bf16(v.x) | ((unsigned int)f32_bf16(v.y) << 16);
    o.y = (unsigned int)f32_bf16(v.z) | ((unsigned int)f32_bf16(v.w) << 16);
    ((v2u*)dst)[j] = o;
  }
}

// ---------------------------------------------------------------------------
// value[b][s][h] f32 -> vt[b][h][s] bf16 (32x32 LDS tile transpose)
// ---------------------------------------------------------------------------
__global__ void tconv_kernel(const float* __restrict__ v,
                             ushort_t* __restrict__ vt) {
  __shared__ float tile[32][33];
  int b  = blockIdx.z;
  int h0 = blockIdx.x * 32, s0 = blockIdx.y * 32;
  int tx = threadIdx.x, ty = threadIdx.y;   // 32 x 8
  const float* src = v  + (long long)b * S1_ * H_;
  ushort_t*    dst = vt + (long long)b * H_  * S1_;
#pragma unroll
  for (int i = 0; i < 4; ++i) {
    int s = s0 + ty + i * 8;
    tile[ty + i * 8][tx] = src[(long long)s * H_ + h0 + tx];
  }
  __syncthreads();
#pragma unroll
  for (int i = 0; i < 4; ++i) {
    int h = h0 + ty + i * 8;
    dst[(long long)h * S1_ + s0 + tx] = f32_bf16(tile[tx][ty + i * 8]);
  }
}

// ---------------------------------------------------------------------------
// Batched GEMM: C[m][n] = sum_k A[m][k]*Bt[n][k] (+bias[n])
// A: [M][K] bf16, Bt: [N][K] bf16. Block tile 128x128x32, 8 waves.
// Each wave: 4x2 tiles of v_wmma_f32_16x16x32_bf16.
// Requires M%128==0, N%128==0, K%32==0 and K/32 >= 3 (true at all call sites).
// ---------------------------------------------------------------------------
#define BM 128
#define BN 128
#define BK 32
#define LDT 40                      // padded LDS row (ushorts)
#define NSTAGE 3
#define STAGE_E (BM * LDT)          // elements per stage (same for A and B)
#define STAGE_B (STAGE_E * 2)       // bytes per stage

union FragU { v4u u[2]; v16bf v; };

template <bool HAS_BIAS, bool OUT_F32, bool OUT_BF16>
__global__ __launch_bounds__(256)
void gemm_bf16_bt(const ushort_t* __restrict__ A, const ushort_t* __restrict__ Bt,
                  const float* __restrict__ bias,
                  float* __restrict__ Cf, ushort_t* __restrict__ Cb,
                  int M, int N, int K,
                  long long strA, long long strB, long long strC) {
  __shared__ ushort_t sA[NSTAGE][STAGE_E];
  __shared__ ushort_t sB[NSTAGE][STAGE_E];

  const int z = blockIdx.z;
  A  += (long long)z * strA;
  Bt += (long long)z * strB;

  const int rowBase = blockIdx.y * BM;
  const int colBase = blockIdx.x * BN;
  const int tid  = threadIdx.x;
  const int wave = tid >> 5, lane = tid & 31;
  const int l16  = lane & 15, half = lane >> 4;
  const int waveM = (wave >> 2) * 64;   // 2 wave-rows * 64
  const int waveN = (wave & 3) * 32;    // 4 wave-cols * 32

  // tile-fill: 512 chunks of 8 ushorts (16B), 2 per thread, per matrix
  const int rA0 = tid >> 2,        kA0 = (tid & 3) * 8;
  const int rA1 = 64 + (tid >> 2), kA1 = kA0;

  // running 64-bit global byte addresses (advance 64B per k-tile)
  unsigned long long ga0 = (unsigned long long)(size_t)(A  + (long long)(rowBase + rA0) * K + kA0);
  unsigned long long ga1 = (unsigned long long)(size_t)(A  + (long long)(rowBase + rA1) * K + kA1);
  unsigned long long gb0 = (unsigned long long)(size_t)(Bt + (long long)(colBase + rA0) * K + kA0);
  unsigned long long gb1 = (unsigned long long)(size_t)(Bt + (long long)(colBase + rA1) * K + kA1);

  // per-thread LDS byte addresses in stage 0 (low 32 bits of generic addr)
  const unsigned la0 = (unsigned)(size_t)&sA[0][rA0 * LDT + kA0];
  const unsigned la1 = (unsigned)(size_t)&sA[0][rA1 * LDT + kA1];
  const unsigned lb0 = (unsigned)(size_t)&sB[0][rA0 * LDT + kA0];
  const unsigned lb1 = (unsigned)(size_t)&sB[0][rA1 * LDT + kA1];

  unsigned bufOff = 0;                 // issue-stage byte offset {0,S,2S}
  unsigned curOff = 0;                 // compute-stage element offset

  // issue one k-tile: 4 async 16B copies per thread -> ASYNCcnt +4 per wave
  auto issue = [&]() {
    unsigned d0 = la0 + bufOff, d1 = la1 + bufOff;
    unsigned d2 = lb0 + bufOff, d3 = lb1 + bufOff;
    asm volatile("global_load_async_to_lds_b128 %0, %1, off" :: "v"(d0), "v"(ga0) : "memory");
    asm volatile("global_load_async_to_lds_b128 %0, %1, off" :: "v"(d1), "v"(ga1) : "memory");
    asm volatile("global_load_async_to_lds_b128 %0, %1, off" :: "v"(d2), "v"(gb0) : "memory");
    asm volatile("global_load_async_to_lds_b128 %0, %1, off" :: "v"(d3), "v"(gb1) : "memory");
    ga0 += BK * 2; ga1 += BK * 2; gb0 += BK * 2; gb1 += BK * 2;
    bufOff = (bufOff == 2u * STAGE_B) ? 0u : bufOff + STAGE_B;
  };

  v8f acc[4][2];
  v8f vzero = {};
#pragma unroll
  for (int i = 0; i < 4; ++i)
#pragma unroll
    for (int j = 0; j < 2; ++j) acc[i][j] = vzero;

  // one k-tile of math: 4 A-frags, 2 B-frags, 8 WMMAs (ISA 7.12.2 layouts)
  auto compute = [&]() {
    const ushort_t* baseA = &sA[0][0] + curOff;
    const ushort_t* baseB = &sB[0][0] + curOff;
    FragU af[4], bfr[2];
#pragma unroll
    for (int i = 0; i < 4; ++i) {
      const ushort_t* p = baseA + (waveM + i * 16 + l16) * LDT + half * 8;
      af[i].u[0] = *(const v4u*)p;          // k = half*8 .. +7
      af[i].u[1] = *(const v4u*)(p + 16);   // k = 16+half*8 .. +7
    }
#pragma unroll
    for (int j = 0; j < 2; ++j) {
      const ushort_t* p = baseB + (waveN + j * 16 + l16) * LDT + half * 8;
      bfr[j].u[0] = *(const v4u*)p;         // k = half*8 .. +7
      bfr[j].u[1] = *(const v4u*)(p + 16);  // k = 16+half*8 .. +7
    }
#pragma unroll
    for (int i = 0; i < 4; ++i)
#pragma unroll
      for (int j = 0; j < 2; ++j)
        acc[i][j] = __builtin_amdgcn_wmma_f32_16x16x32_bf16(
            false, af[i].v, false, bfr[j].v, (short)0, acc[i][j], false, false);
    curOff = (curOff == 2u * STAGE_E) ? 0u : curOff + STAGE_E;
  };

  const int kt_max = K / BK;           // >= 3 at all call sites

  // prologue: fill 2 pipeline stages
  issue();                                              // tile 0 -> stage 0
  issue();                                              // tile 1 -> stage 1
  asm volatile("s_wait_asynccnt 0x4" ::: "memory");     // tile 0 resident
  __syncthreads();

  // steady state: always issue (straight-line, no modulo, no branches)
  for (int kt = 0; kt < kt_max - 2; ++kt) {
    issue();                                            // tile kt+2
    compute();                                          // tile kt
    asm volatile("s_wait_asynccnt 0x4" ::: "memory");   // tile kt+1 resident
    __syncthreads();
  }
  // drain: last two tiles
  compute();                                            // tile kt_max-2
  asm volatile("s_wait_asynccnt 0x0" ::: "memory");     // tile kt_max-1 resident
  __syncthreads();
  compute();                                            // tile kt_max-1

  // epilogue: C/D layout -> m = half*8 + r, n = l16 (coalesced over lanes)
  float*    Cfb = OUT_F32  ? (Cf + (long long)z * strC) : nullptr;
  ushort_t* Cbb = OUT_BF16 ? (Cb + (long long)z * strC) : nullptr;
#pragma unroll
  for (int j = 0; j < 2; ++j) {
    const int nn = colBase + waveN + j * 16 + l16;
    float bv = 0.0f;
    if (HAS_BIAS) bv = bias[nn];
#pragma unroll
    for (int i = 0; i < 4; ++i) {
#pragma unroll
      for (int r = 0; r < 8; ++r) {
        const int mm = rowBase + waveM + i * 16 + half * 8 + r;
        const float val = acc[i][j][r] + bv;
        const long long off = (long long)mm * N + nn;
        if (OUT_F32)  Cfb[off] = val;
        if (OUT_BF16) Cbb[off] = f32_bf16(val);
      }
    }
  }
}

// ---------------------------------------------------------------------------
// Masked softmax over contiguous rows of scoresT [B*S2][S1], in place.
// wave32 butterfly reductions + one 8-slot LDS exchange per phase.
// mask[b][s1][s2]==true -> -inf before softmax. Also emits bf16 copy.
// ---------------------------------------------------------------------------
__device__ __forceinline__ float wred_max(float x) {
#pragma unroll
  for (int o = 16; o > 0; o >>= 1) x = fmaxf(x, __shfl_xor(x, o, 32));
  return x;
}
__device__ __forceinline__ float wred_sum(float x) {
#pragma unroll
  for (int o = 16; o > 0; o >>= 1) x += __shfl_xor(x, o, 32);
  return x;
}

__global__ __launch_bounds__(256)
void softmax_mask_kernel(float* __restrict__ scores, ushort_t* __restrict__ w16,
                         const unsigned char* __restrict__ mask) {
  __shared__ float redm[8];
  __shared__ float reds[8];
  const int rid = blockIdx.x;          // b*S2 + s2
  const int b   = rid >> 11;
  const int k   = rid & (S2_ - 1);
  float*      row  = scores + (long long)rid * S1_;
  ushort_t*   wrow = w16    + (long long)rid * S1_;
  const unsigned char* mcol = mask + (long long)b * S1_ * S2_ + k;
  const int tid  = threadIdx.x;
  const int wave = tid >> 5, lane = tid & 31;

  float v[8];
  float mx = -__builtin_inff();
#pragma unroll
  for (int u = 0; u < 8; ++u) {
    const int s = u * 256 + tid;
    float x = row[s];
    if (mcol[(long long)s * S2_]) x = -__builtin_inff();
    v[u] = x;
    mx = fmaxf(mx, x);
  }
  mx = wred_max(mx);
  if (lane == 0) redm[wave] = mx;
  __syncthreads();
  float M = redm[0];
#pragma unroll
  for (int w = 1; w < 8; ++w) M = fmaxf(M, redm[w]);

  float sum = 0.0f;
#pragma unroll
  for (int u = 0; u < 8; ++u) { v[u] = __expf(v[u] - M); sum += v[u]; }
  sum = wred_sum(sum);
  if (lane == 0) reds[wave] = sum;
  __syncthreads();
  float S = reds[0];
#pragma unroll
  for (int w = 1; w < 8; ++w) S += reds[w];

  const float inv = 1.0f / S;
#pragma unroll
  for (int u = 0; u < 8; ++u) {
    const int s = u * 256 + tid;
    const float w = v[u] * inv;
    row[s]  = w;
    wrow[s] = f32_bf16(w);
  }
}

// ---------------------------------------------------------------------------
extern "C" void kernel_launch(void* const* d_in, const int* in_sizes, int n_in,
                              void* d_out, int out_size, void* d_ws, size_t ws_size,
                              hipStream_t stream) {
  const float*         value = (const float*)d_in[0];
  const float*         keyp  = (const float*)d_in[1];
  const float*         query = (const float*)d_in[2];
  const unsigned char* mask  = (const unsigned char*)d_in[3];
  const float*         W     = (const float*)d_in[4];
  const float*         bias  = (const float*)d_in[5];

  float* ctx  = (float*)d_out;                              // [B,S2,H]
  float* wout = (float*)d_out + (size_t)B_ * S2_ * H_;      // [B,S2,S1]

  char* ws = (char*)d_ws;
  size_t off = 0;
  const size_t nBSH = (size_t)B_ * S1_ * H_;
  ushort_t* q16  = (ushort_t*)(ws + off); off += nBSH * 2;                 // Q bf16
  ushort_t* k16  = (ushort_t*)(ws + off); off += nBSH * 2;                 // K bf16
  ushort_t* vt16 = (ushort_t*)(ws + off); off += nBSH * 2;                 // V^T bf16
  ushort_t* w16m = (ushort_t*)(ws + off); off += (size_t)H_ * H_ * 2;      // W bf16
  ushort_t* qp16 = (ushort_t*)(ws + off); off += nBSH * 2;                 // q_proj bf16
  ushort_t* wt16 = (ushort_t*)(ws + off); off += (size_t)B_ * S2_ * S1_ * 2; // weights bf16

  // 0) precision conversion / layout prep
  cvt_f32_bf16_kernel<<<1024, 256, 0, stream>>>(query, q16, (long long)(nBSH / 4));
  cvt_f32_bf16_kernel<<<1024, 256, 0, stream>>>(keyp,  k16, (long long)(nBSH / 4));
  cvt_f32_bf16_kernel<<<64,   256, 0, stream>>>(W,     w16m, (long long)((size_t)H_ * H_ / 4));
  {
    dim3 g(H_ / 32, S1_ / 32, B_), b(32, 8);
    tconv_kernel<<<g, b, 0, stream>>>(value, vt16);
  }

  // 1) q_proj = Q @ W^T + b  -> bf16   (M=S1, N=H, K=H)
  {
    dim3 g(H_ / BN, S1_ / BM, B_);
    gemm_bf16_bt<true, false, true><<<g, 256, 0, stream>>>(q16, w16m, bias, nullptr, qp16,
        S1_, H_, H_, (long long)S1_ * H_, 0LL, (long long)S1_ * H_);
  }
  // 2) scoresT = K @ q_proj^T -> fp32 into d_out weights region (M=S2, N=S1, K=H)
  {
    dim3 g(S1_ / BN, S2_ / BM, B_);
    gemm_bf16_bt<false, true, false><<<g, 256, 0, stream>>>(k16, qp16, nullptr, wout, nullptr,
        S2_, S1_, H_, (long long)S2_ * H_, (long long)S1_ * H_, (long long)S2_ * S1_);
  }
  // 3) masked softmax over S1 rows, in place; bf16 copy for GEMM3
  softmax_mask_kernel<<<B_ * S2_, 256, 0, stream>>>(wout, wt16, mask);

  // 4) context = weights @ V -> fp32 (M=S2, N=H, K=S1; Bt = V^T[h][s])
  {
    dim3 g(H_ / BN, S2_ / BM, B_);
    gemm_bf16_bt<false, true, false><<<g, 256, 0, stream>>>(wt16, vt16, nullptr, ctx, nullptr,
        S2_, H_, S1_, (long long)S2_ * S1_, (long long)H_ * S1_, (long long)S2_ * H_);
  }
}
